// MultiHeadAttention_83571473645732
// MI455X (gfx1250) — matrix operand order repaired and checked by reading, not compiled, not run on hardware
//
#include <hip/hip_runtime.h>
#include <hip/hip_bf16.h>

// ---------------------------------------------------------------------------
// MI455X / gfx1250 fused MHA forward:
//   1) cvt x, W_qkv to f16
//   2) QKV GEMM: v_wmma_f32_16x16x32_f16, double-buffered LDS W tile,
//      B-fragments in distinct regs so WMMAs issue back-to-back.
//   3) fused flash-attention: 64-key blocks, LDS K/V^T tiles shared by 8
//      waves with register-held early loads hiding stage latency.
// ---------------------------------------------------------------------------

typedef __attribute__((ext_vector_type(8)))  _Float16 v8h;
typedef __attribute__((ext_vector_type(16))) _Float16 v16h;
typedef __attribute__((ext_vector_type(8)))  float    v8f;

#define D_MODEL 1024
#define N_HEADS 8
#define DH      128
#define SEQ     2048
#define BATCH   2
#define M_TOK   (BATCH * SEQ)     // 4096 tokens
#define N_QKV   (3 * D_MODEL)     // 3072 output features

// LDS row strides (halves) chosen so 16 consecutive rows map to distinct banks
#define K_STRIDE  136             // 64x128 K tile rows   (n*68 dw -> n*4 banks)
#define V_STRIDE  72              // 128x64 V^T tile rows (n*36 mod 64 distinct)
#define P_STRIDE  72              // 16x64 P tile rows
#define W_STRIDE  40              // 64x32 W tile rows    (n*20 mod 64 distinct)

// ---- fragment loaders (wave32 layouts per CDNA5 ISA) -----------

// A-matrix 16x32 f16: lane&15 = row M; g = lane>>4 selects half-chunks
// [g*8 .. g*8+7] and [g*8+16 .. g*8+23] of the 32-wide K window.
__device__ __forceinline__ v16h load_a_frag(const _Float16* row, int g) {
  const v8h c0 = *reinterpret_cast<const v8h*>(row + g * 8);
  const v8h c1 = *reinterpret_cast<const v8h*>(row + g * 8 + 16);
  v16h r;
#pragma unroll
  for (int i = 0; i < 8; ++i) { r[i] = c0[i]; r[i + 8] = c1[i]; }
  return r;
}

// B-matrix 32x16 f16 (column-major storage Bt[n][k]): lane&15 = column N,
// g = lane>>4 selects K chunks [g*8 .. g*8+7] and [g*8+16 .. g*8+23] (same as A),
// loaded as 2x16B so only 16B alignment is required (padded LDS rows).
__device__ __forceinline__ v16h load_b_frag(const _Float16* row, int g) {
  const v8h c0 = *reinterpret_cast<const v8h*>(row + g * 8);
  const v8h c1 = *reinterpret_cast<const v8h*>(row + g * 8 + 16);
  v16h r;
#pragma unroll
  for (int i = 0; i < 8; ++i) { r[i] = c0[i]; r[i + 8] = c1[i]; }
  return r;
}

__device__ __forceinline__ v8f wmma_f16(v16h a, v16h b, v8f c) {
  return __builtin_amdgcn_wmma_f32_16x16x32_f16(false, a, false, b,
                                                (short)0, c, false, false);
}

// ---- stage 1: f32 -> f16 conversion ---------------------------------------

__global__ void __launch_bounds__(256) cvt_f32_f16_kernel(
    const float* __restrict__ in, _Float16* __restrict__ out, int n) {
  int i = blockIdx.x * 256 + threadIdx.x;
  if (i < n) out[i] = (_Float16)in[i];
}

// ---- stage 2: QKV projection GEMM -----------------------------------------
// Block = 8 waves: shared 64-feature N strip (double-buffered LDS W tile),
// per-wave 16-row M tile => block computes 128x64. K-loop of 32.

__global__ void __launch_bounds__(256) qkv_gemm_kernel(
    const _Float16* __restrict__ xh, const _Float16* __restrict__ wh,
    const float* __restrict__ bias,
    _Float16* __restrict__ qws, _Float16* __restrict__ kws,
    _Float16* __restrict__ vtws) {
  __shared__ _Float16 Wl[2][64 * W_STRIDE];       // 2 x 5.1 KB padded W tiles

  const int tid  = threadIdx.x;
  const int lane = tid & 31;
  const int wv   = tid >> 5;
  const int g    = lane >> 4;
  const int l15  = lane & 15;

  const int nStrips = N_QKV / 64;                 // 48
  const int mBlk   = blockIdx.x / nStrips;        // 0..31
  const int nStrip = blockIdx.x % nStrips;
  const int row0 = mBlk * 128 + wv * 16;
  const int col0 = nStrip * 64;

  const _Float16* arow = xh + (size_t)(row0 + l15) * D_MODEL;

  // staging: thread -> (row r = tid/4, 16B chunk off = (tid%4)*8 halves)
  const int sr   = tid >> 2;
  const int soff = (tid & 3) * 8;
  const _Float16* wsrc = wh + (size_t)(col0 + sr) * D_MODEL + soff;
  const int woff = sr * W_STRIDE + soff;

  // prologue: stage k0 = 0 into buffer 0
  *reinterpret_cast<v8h*>(&Wl[0][woff]) = *reinterpret_cast<const v8h*>(wsrc);
  __syncthreads();

  v8f acc[4] = {};
  for (int ki = 0; ki < 32; ++ki) {
    const int k0  = ki * 32;
    const int cur = ki & 1;
    // issue next tile's global load early (clamped reload on last iter)
    const int kn = (ki < 31) ? (k0 + 32) : k0;
    const v8h wn = *reinterpret_cast<const v8h*>(wsrc + kn);

    // compute: distinct B-frag regs -> one dscnt wait, 4 back-to-back WMMAs
    v16h a  = load_a_frag(arow + k0, g);
    v16h b0 = load_b_frag(&Wl[cur][( 0 + l15) * W_STRIDE], g);
    v16h b1 = load_b_frag(&Wl[cur][(16 + l15) * W_STRIDE], g);
    v16h b2 = load_b_frag(&Wl[cur][(32 + l15) * W_STRIDE], g);
    v16h b3 = load_b_frag(&Wl[cur][(48 + l15) * W_STRIDE], g);
    acc[0] = wmma_f16(a, b0, acc[0]);
    acc[1] = wmma_f16(a, b1, acc[1]);
    acc[2] = wmma_f16(a, b2, acc[2]);
    acc[3] = wmma_f16(a, b3, acc[3]);

    // stage into the other buffer (read last iteration, free to overwrite)
    *reinterpret_cast<v8h*>(&Wl[cur ^ 1][woff]) = wn;
    __syncthreads();
  }

#pragma unroll
  for (int t = 0; t < 4; ++t) {
#pragma unroll
    for (int v = 0; v < 8; ++v) {
      const int m     = v + g * 8;                 // C/D row mapping
      const int token = row0 + m;
      const int f     = col0 + t * 16 + l15;       // C/D col = lane&15
      const float val = acc[t][v] + bias[f];
      const int which = f >> 10;                   // 0=Q 1=K 2=V
      const int d  = f & (D_MODEL - 1);
      const int h  = d >> 7;
      const int dh = d & (DH - 1);
      const int b  = token >> 11;
      const int s  = token & (SEQ - 1);
      const _Float16 hv = (_Float16)val;
      const size_t head = (size_t)(b * N_HEADS + h);
      if (which == 0)      qws[(head * SEQ + s) * DH + dh]  = hv;
      else if (which == 1) kws[(head * SEQ + s) * DH + dh]  = hv;
      else                 vtws[(head * DH + dh) * SEQ + s] = hv;
    }
  }
}

// ---- stage 3: fused flash attention ---------------------------------------
// Block = 8 waves, same (b,h); 64-key iterations, K/V^T tiles in LDS.
// Next block's tiles are loaded into registers BEFORE compute so the global
// latency hides behind ~32 WMMAs; stores + barriers happen after compute.

struct StageRegs { v8h kc[4]; v8h vc[4]; };

__device__ __forceinline__ StageRegs stage_load(
    const _Float16* __restrict__ K, const _Float16* __restrict__ Vt,
    int kn, int tid) {
  StageRegs s;
#pragma unroll
  for (int c = 0; c < 4; ++c) {
    const int ck = tid + c * 256;                  // 1024 chunks of 8 halves
    const int kr = ck >> 4, ko = (ck & 15) * 8;
    s.kc[c] = *reinterpret_cast<const v8h*>(K + (size_t)(kn + kr) * DH + ko);
    const int vr = ck >> 3, vo = (ck & 7) * 8;
    s.vc[c] = *reinterpret_cast<const v8h*>(Vt + (size_t)vr * SEQ + kn + vo);
  }
  return s;
}

__device__ __forceinline__ void stage_store(
    const StageRegs& s, _Float16* Kl, _Float16* Vtl, int tid) {
#pragma unroll
  for (int c = 0; c < 4; ++c) {
    const int ck = tid + c * 256;
    const int kr = ck >> 4, ko = (ck & 15) * 8;
    *reinterpret_cast<v8h*>(&Kl[kr * K_STRIDE + ko]) = s.kc[c];
    const int vr = ck >> 3, vo = (ck & 7) * 8;
    *reinterpret_cast<v8h*>(&Vtl[vr * V_STRIDE + vo]) = s.vc[c];
  }
}

__global__ void __launch_bounds__(256) flash_attn_kernel(
    const _Float16* __restrict__ qws, const _Float16* __restrict__ kws,
    const _Float16* __restrict__ vtws, float* __restrict__ out) {
  __shared__ _Float16 Kl[64 * K_STRIDE];          // 17.0 KB
  __shared__ _Float16 Vtl[128 * V_STRIDE];        // 18.0 KB
  __shared__ _Float16 Pl[8][16 * P_STRIDE];       // 18.0 KB (wave-private P)

  const int tid  = threadIdx.x;
  const int lane = tid & 31;
  const int wv   = tid >> 5;
  const int g    = lane >> 4;
  const int l15  = lane & 15;

  const int qgroups = SEQ / 128;                  // 16
  const int bh = blockIdx.x / qgroups;
  const int qg = blockIdx.x % qgroups;
  const int b  = bh >> 3;
  const int h  = bh & (N_HEADS - 1);
  const int q0 = qg * 128 + wv * 16;

  const size_t head = (size_t)(b * N_HEADS + h);
  const _Float16* Q  = qws  + head * SEQ * DH;
  const _Float16* K  = kws  + head * SEQ * DH;
  const _Float16* Vt = vtws + head * DH * SEQ;

  // preload the 16x128 query tile as 4 A-fragments (K-steps of 32)
  const _Float16* qrow = Q + (size_t)(q0 + l15) * DH;
  v16h qf[4];
#pragma unroll
  for (int i = 0; i < 4; ++i) qf[i] = load_a_frag(qrow + 32 * i, g);

  v8f o[8] = {};
  float mrow[8], lrow[8];
#pragma unroll
  for (int v = 0; v < 8; ++v) { mrow[v] = -1.0e30f; lrow[v] = 0.0f; }

  const float scale = 0.0883883476483184f;        // 1/sqrt(128)
  _Float16* myP = &Pl[wv][0];

  // prologue: stage key-block 0
  {
    StageRegs s0 = stage_load(K, Vt, 0, tid);
    stage_store(s0, Kl, Vtl, tid);
  }
  __syncthreads();

  for (int kb = 0; kb < SEQ; kb += 64) {
    // issue next block's global loads now; they land during compute
    const int kn = (kb + 64 < SEQ) ? (kb + 64) : kb;
    StageRegs st = stage_load(K, Vt, kn, tid);

    // ---- scores: four 16x16 tiles over 64 keys, f32 accum ----
    v8f sc[4];
#pragma unroll
    for (int kt = 0; kt < 4; ++kt) {
      const _Float16* kr = &Kl[(kt * 16 + l15) * K_STRIDE];
      v16h b0 = load_b_frag(kr +  0, g);
      v16h b1 = load_b_frag(kr + 32, g);
      v16h b2 = load_b_frag(kr + 64, g);
      v16h b3 = load_b_frag(kr + 96, g);
      v8f s = {};
      s = wmma_f16(qf[0], b0, s);
      s = wmma_f16(qf[1], b1, s);
      s = wmma_f16(qf[2], b2, s);
      s = wmma_f16(qf[3], b3, s);
      sc[kt] = s;
    }

    // ---- online softmax (row r = v + g*8 of each half-wave) ----
#pragma unroll
    for (int v = 0; v < 8; ++v) {
      float a0 = sc[0][v] * scale, a1 = sc[1][v] * scale;
      float a2 = sc[2][v] * scale, a3 = sc[3][v] * scale;
      float mx = fmaxf(fmaxf(a0, a1), fmaxf(a2, a3));
      mx = fmaxf(mx, __shfl_xor(mx, 1, 32));
      mx = fmaxf(mx, __shfl_xor(mx, 2, 32));
      mx = fmaxf(mx, __shfl_xor(mx, 4, 32));
      mx = fmaxf(mx, __shfl_xor(mx, 8, 32));
      const float mnew = fmaxf(mrow[v], mx);
      const float p0 = __expf(a0 - mnew), p1 = __expf(a1 - mnew);
      const float p2 = __expf(a2 - mnew), p3 = __expf(a3 - mnew);
      float rs = (p0 + p1) + (p2 + p3);
      rs += __shfl_xor(rs, 1, 32);
      rs += __shfl_xor(rs, 2, 32);
      rs += __shfl_xor(rs, 4, 32);
      rs += __shfl_xor(rs, 8, 32);
      const float alpha = __expf(mrow[v] - mnew);
      lrow[v] = lrow[v] * alpha + rs;
      mrow[v] = mnew;
#pragma unroll
      for (int t = 0; t < 8; ++t) o[t][v] *= alpha;
      const int r = v + g * 8;                     // stage P row-major, padded
      myP[r * P_STRIDE +  0 + l15] = (_Float16)p0;
      myP[r * P_STRIDE + 16 + l15] = (_Float16)p1;
      myP[r * P_STRIDE + 32 + l15] = (_Float16)p2;
      myP[r * P_STRIDE + 48 + l15] = (_Float16)p3;
    }
    asm volatile("s_wait_dscnt 0x0" ::: "memory"); // wave-private P ready

    // ---- P(16x64) @ V(64x128): P as two A-frags, V^T rows as B ----
    v16h pf0 = load_a_frag(myP + l15 * P_STRIDE, g);
    v16h pf1 = load_a_frag(myP + l15 * P_STRIDE + 32, g);
#pragma unroll
    for (int t = 0; t < 8; t += 2) {
      const _Float16* vr0 = &Vtl[((t + 0) * 16 + l15) * V_STRIDE];
      const _Float16* vr1 = &Vtl[((t + 1) * 16 + l15) * V_STRIDE];
      v16h vf00 = load_b_frag(vr0,      g);
      v16h vf01 = load_b_frag(vr0 + 32, g);
      v16h vf10 = load_b_frag(vr1,      g);
      v16h vf11 = load_b_frag(vr1 + 32, g);
      o[t + 0] = wmma_f16(pf0, vf00, o[t + 0]);
      o[t + 0] = wmma_f16(pf1, vf01, o[t + 0]);
      o[t + 1] = wmma_f16(pf0, vf10, o[t + 1]);
      o[t + 1] = wmma_f16(pf1, vf11, o[t + 1]);
    }

    __syncthreads();                               // all waves done with tiles
    stage_store(st, Kl, Vtl, tid);                 // loads landed during compute
    __syncthreads();                               // next tiles visible
  }

  // ---- finalize: divide by row sums, write [B, S, H*Dh] f32 ----
  float inv[8];
#pragma unroll
  for (int v = 0; v < 8; ++v) inv[v] = 1.0f / lrow[v];
#pragma unroll
  for (int t = 0; t < 8; ++t) {
#pragma unroll
    for (int v = 0; v < 8; ++v) {
      const int r  = v + g * 8;
      const int q  = q0 + r;
      const int dh = t * 16 + l15;
      out[((size_t)(b * SEQ + q)) * D_MODEL + h * DH + dh] = o[t][v] * inv[v];
    }
  }
}

// ---------------------------------------------------------------------------

extern "C" void kernel_launch(void* const* d_in, const int* in_sizes, int n_in,
                              void* d_out, int out_size, void* d_ws, size_t ws_size,
                              hipStream_t stream) {
  (void)in_sizes; (void)n_in; (void)out_size; (void)ws_size;

  const float* x    = (const float*)d_in[0];   // [2, 2048, 1024] f32
  const float* W    = (const float*)d_in[1];   // [3072, 1024] f32 (row = feature)
  const float* bias = (const float*)d_in[2];   // [3072] f32
  float* out = (float*)d_out;                  // [2, 2048, 1024] f32

  // workspace carve-up (f16): xh | wh | Q | K | Vt  -> ~38 MB total
  _Float16* xh   = (_Float16*)d_ws;
  _Float16* wh   = xh  + (size_t)M_TOK * D_MODEL;                 // 4,194,304
  _Float16* qws  = wh  + (size_t)N_QKV * D_MODEL;                 // 3,145,728
  _Float16* kws  = qws + (size_t)BATCH * N_HEADS * SEQ * DH;      // 4,194,304
  _Float16* vtws = kws + (size_t)BATCH * N_HEADS * SEQ * DH;      // 4,194,304

  const int nx = M_TOK * D_MODEL;   // 4,194,304 (multiple of 256)
  const int nw = N_QKV * D_MODEL;   // 3,145,728 (multiple of 256)
  hipLaunchKernelGGL(cvt_f32_f16_kernel, dim3(nx / 256), dim3(256), 0, stream, x, xh, nx);
  hipLaunchKernelGGL(cvt_f32_f16_kernel, dim3(nw / 256), dim3(256), 0, stream, W, wh, nw);

  // GEMM: 32 M-blocks (128 rows) x 48 N-strips (64 feats) = 1536 blocks
  hipLaunchKernelGGL(qkv_gemm_kernel, dim3(1536), dim3(256), 0, stream,
                     xh, wh, bias, qws, kws, vtws);

  // attention: 2*8 heads * 16 query-groups (128 q each) = 256 blocks
  hipLaunchKernelGGL(flash_attn_kernel, dim3(BATCH * N_HEADS * (SEQ / 128)),
                     dim3(256), 0, stream, qws, kws, vtws, out);
}
